// SRNN_66666482368616
// MI455X (gfx1250) — matrix operand order repaired and checked by reading, not compiled, not run on hardware
//
#include <hip/hip_runtime.h>
#include <math.h>

// ---------------------------------------------------------------------------
// SRNN (VRNN) forward loss for MI455X / gfx1250, wave32 + WMMA bf16.
// Round 2: bf16-resident weights & GEMM activations (one-time conversion),
// fp32 accumulate + fp32 elementwise math. Inner GEMM loops are pure
// global_load_b128(bf16) -> v_wmma_f32_16x16x32_bf16.
//
// Input flattening assumption (setup_inputs dict insertion order, nested
// tuples flattened in order):
//   0:x 1:x_sl 2:eps 3:emb_W 4:emb_b 5:gruf_Wi 6:gruf_Wh 7:gruf_bi 8:gruf_bh
//   9:grub_Wi 10:grub_Wh 11:grub_bi 12:grub_bh
//   13..15:prior_Ws 16..18:prior_bs 19:prior_mu_W 20:prior_mu_b
//   21:prior_sd_W 22:prior_sd_b
//   23..25:enc_Ws 26..28:enc_bs 29:enc_mu_W 30:enc_mu_b 31:enc_sd_W 32:enc_sd_b
//   33..35:dec_Ws 36..38:dec_bs 39:lik_mu_W 40:lik_mu_b 41:lik_sd_W 42:lik_sd_b
// ---------------------------------------------------------------------------

#define B_   64
#define T_   512
#define DX_  128
#define H_   512
#define Z_   128
#define G_   (3 * H_)              // 1536 GRU gate width
#define NBLK_SCAN 32               // persistent blocks for sequential scans
#define NROWTILE  (B_ * T_ / 16)   // 2048 row tiles for decoder/emb

typedef __bf16 bf16_t;
typedef __attribute__((ext_vector_type(16))) __bf16 v16bf;
typedef __attribute__((ext_vector_type(8)))  __bf16 v8bf;
typedef __attribute__((ext_vector_type(8)))  float  v8f;

__device__ __forceinline__ v8f v8f_zero() {
  v8f z;
#pragma unroll
  for (int i = 0; i < 8; ++i) z[i] = 0.0f;
  return z;
}

__device__ __forceinline__ float sigf(float x)      { return 1.0f / (1.0f + __expf(-x)); }
__device__ __forceinline__ float softplusf_(float x){ return x > 20.0f ? x : log1pf(__expf(x)); }
__device__ __forceinline__ float reluf(float x)     { return fmaxf(x, 0.0f); }

__device__ __forceinline__ v16bf cat8(v8bf lo, v8bf hi) {
  return __builtin_shufflevector(lo, hi, 0, 1, 2, 3, 4, 5, 6, 7,
                                 8, 9, 10, 11, 12, 13, 14, 15);
}

// ---------------------------------------------------------------------------
// 16x16 f32 tile accumulation: acc += A(16xK) * W(16 rows x K cols)^T.
// Both operands bf16 in memory. VGPR layouts per CDNA5 ISA 7.12.2:
//   A 16x32 bf16 : lane half h holds k in [8h, 8h+8) and [16+8h, 24+8h)
//   B 32x16 bf16 : lane half h holds k in [8h, 8h+8) and [16+8h, 24+8h)
//   C/D 16x16 f32: acc[r] -> row (r + 8h), col (lane & 15)
// Arow/Wrow are this lane's row base pointers; 16-byte aligned; K % 32 == 0.
// ---------------------------------------------------------------------------
__device__ __forceinline__ v8f wmma_kloop_bb(const bf16_t* __restrict__ Arow,
                                             const bf16_t* __restrict__ Wrow,
                                             int K, v8f acc, int half) {
  for (int k0 = 0; k0 < K; k0 += 32) {
    v8bf a0 = *(const v8bf*)(Arow + k0 + 8 * half);
    v8bf a1 = *(const v8bf*)(Arow + k0 + 16 + 8 * half);
    v8bf b0 = *(const v8bf*)(Wrow + k0 + 8 * half);
    v8bf b1 = *(const v8bf*)(Wrow + k0 + 16 + 8 * half);
    acc = __builtin_amdgcn_wmma_f32_16x16x32_bf16(false, cat8(a0, a1), false,
                                                  cat8(b0, b1), (short)0, acc,
                                                  false, false);
  }
  return acc;
}

// Variant with fp32 A (in-register convert) and bf16 W — used only by the
// one-shot embedding GEMM whose A operand is the external fp32 input x.
__device__ __forceinline__ v8f wmma_kloop_fb(const float* __restrict__ Arow,
                                             const bf16_t* __restrict__ Wrow,
                                             int K, v8f acc, int half) {
  for (int k0 = 0; k0 < K; k0 += 32) {
    v16bf av;
#pragma unroll
    for (int v = 0; v < 4; ++v) {
      int ka = k0 + 8 * half + 2 * v;
      av[2 * v + 0]     = (bf16_t)Arow[ka + 0];
      av[2 * v + 1]     = (bf16_t)Arow[ka + 1];
      av[8 + 2 * v + 0] = (bf16_t)Arow[ka + 16];
      av[8 + 2 * v + 1] = (bf16_t)Arow[ka + 17];
    }
    v8bf b0 = *(const v8bf*)(Wrow + k0 + 8 * half);
    v8bf b1 = *(const v8bf*)(Wrow + k0 + 16 + 8 * half);
    acc = __builtin_amdgcn_wmma_f32_16x16x32_bf16(false, av, false,
                                                  cat8(b0, b1), (short)0, acc,
                                                  false, false);
  }
  return acc;
}

// Device-scope barrier for persistent cooperative kernels. Counter is zeroed
// by hipMemsetAsync before launch; targets increase monotonically.
__device__ __forceinline__ void grid_sync(unsigned* bar, unsigned target) {
  __syncthreads();
  if (threadIdx.x == 0) {
    __threadfence();
    atomicAdd(bar, 1u);
    volatile unsigned* vb = (volatile unsigned*)bar;
    while (*vb < target) __builtin_amdgcn_s_sleep(2);
    __threadfence();
  }
  __syncthreads();
}

// ---------------------------------------------------------------------------
// Kernel 0: one-time fp32 -> bf16 conversion (weights).
// ---------------------------------------------------------------------------
__global__ __launch_bounds__(256) void f2bf_kernel(const float* __restrict__ s,
                                                   bf16_t* __restrict__ d,
                                                   int n) {
  int i = blockIdx.x * blockDim.x + threadIdx.x;
  if (i < n) d[i] = (bf16_t)s[i];
}

// ---------------------------------------------------------------------------
// Kernel 1: embedding  xe[t][b][:] = x[b][t][:] @ emb_W^T + emb_b  (bf16 out)
// ---------------------------------------------------------------------------
__global__ __launch_bounds__(256) void emb_kernel(const float* __restrict__ x,
                                                  const bf16_t* __restrict__ W,
                                                  const float* __restrict__ bias,
                                                  bf16_t* __restrict__ xe) {
  const int lane = threadIdx.x & 31, half = lane >> 4, l15 = lane & 15;
  const int tile = blockIdx.x * 8 + (threadIdx.x >> 5);
  const int tm = (tile >> 3) * 16;   // rows in [t][b] order
  const int tn = (tile & 7) * 16;    // DX cols
  const int m = tm + l15;
  const int t = m >> 6, b = m & 63;
  const int n = tn + l15;
  v8f acc = wmma_kloop_fb(x + ((size_t)b * T_ + t) * DX_, W + (size_t)n * DX_,
                          DX_, v8f_zero(), half);
#pragma unroll
  for (int r = 0; r < 8; ++r) {
    int row = tm + r + 8 * half, col = tn + l15;
    xe[(size_t)row * DX_ + col] = (bf16_t)(acc[r] + bias[col]);
  }
}

// Shared GRU gate update: reads gi/gh, updates fp32 h + bf16 shadow, writes
// the step output (bf16) for downstream GEMMs.
__device__ __forceinline__ void gru_gate_phase(const float* __restrict__ gi,
                                               const float* __restrict__ gh,
                                               float* __restrict__ h,
                                               bf16_t* __restrict__ hbf,
                                               bf16_t* __restrict__ out_t,
                                               int gtid, int nthr) {
  for (int idx = gtid; idx < B_ * H_; idx += nthr) {
    int b = idx >> 9, j = idx & (H_ - 1);
    size_t o = (size_t)b * G_ + j;
    float r  = sigf(gi[o] + gh[o]);
    float zg = sigf(gi[o + H_] + gh[o + H_]);
    float nn = tanhf(gi[o + 2 * H_] + r * gh[o + 2 * H_]);
    float hnew = (1.0f - zg) * nn + zg * h[idx];
    h[idx] = hnew;
    hbf[idx] = (bf16_t)hnew;
    out_t[idx] = (bf16_t)hnew;
  }
}

// ---------------------------------------------------------------------------
// Kernel 2: forward GRU (persistent, grid-synced).
// step t: gi = u_t @ Wi^T + bi (u_t = xe[t-1], u_0 = 0); gh = h @ Wh^T + bh
// ---------------------------------------------------------------------------
__global__ __launch_bounds__(256) void gru_fwd_kernel(
    const bf16_t* __restrict__ xe, const bf16_t* __restrict__ Wi,
    const bf16_t* __restrict__ Wh, const float* __restrict__ bi,
    const float* __restrict__ bh, bf16_t* __restrict__ dbuf,
    float* __restrict__ h, bf16_t* __restrict__ hbf, float* __restrict__ gi,
    float* __restrict__ gh, unsigned* __restrict__ bar) {
  const int lane = threadIdx.x & 31, half = lane >> 4, l15 = lane & 15;
  const int gw = blockIdx.x * (blockDim.x >> 5) + (threadIdx.x >> 5);
  const int nw = gridDim.x * (blockDim.x >> 5);
  const int gtid = blockIdx.x * blockDim.x + threadIdx.x;
  const int nthr = gridDim.x * blockDim.x;
  unsigned rnd = 0;
  for (int t = 0; t < T_; ++t) {
    const bf16_t* u = xe + (size_t)(t > 0 ? t - 1 : 0) * B_ * DX_;
    for (int tile = gw; tile < 4 * (G_ / 16); tile += nw) {
      const int tm = (tile & 3) * 16, tn = (tile >> 2) * 16;
      const int m = tm + l15, n = tn + l15;
      v8f ai = v8f_zero();
      if (t > 0)
        ai = wmma_kloop_bb(u + (size_t)m * DX_, Wi + (size_t)n * DX_, DX_, ai,
                           half);
      v8f ah = wmma_kloop_bb(hbf + (size_t)m * H_, Wh + (size_t)n * H_, H_,
                             v8f_zero(), half);
#pragma unroll
      for (int r = 0; r < 8; ++r) {
        int row = tm + r + 8 * half, col = tn + l15;
        gi[(size_t)row * G_ + col] = ai[r] + bi[col];
        gh[(size_t)row * G_ + col] = ah[r] + bh[col];
      }
    }
    grid_sync(bar, (++rnd) * gridDim.x);
    gru_gate_phase(gi, gh, h, hbf, dbuf + (size_t)t * B_ * H_, gtid, nthr);
    grid_sync(bar, (++rnd) * gridDim.x);
  }
}

// ---------------------------------------------------------------------------
// Kernel 3: backward GRU over reversed concat([xe, d]) (input dim 640)
// ---------------------------------------------------------------------------
__global__ __launch_bounds__(256) void gru_bwd_kernel(
    const bf16_t* __restrict__ xe, const bf16_t* __restrict__ dbuf,
    const bf16_t* __restrict__ Wi, const bf16_t* __restrict__ Wh,
    const float* __restrict__ bi, const float* __restrict__ bh,
    bf16_t* __restrict__ abuf, float* __restrict__ h,
    bf16_t* __restrict__ hbf, float* __restrict__ gi, float* __restrict__ gh,
    unsigned* __restrict__ bar) {
  const int lane = threadIdx.x & 31, half = lane >> 4, l15 = lane & 15;
  const int gw = blockIdx.x * (blockDim.x >> 5) + (threadIdx.x >> 5);
  const int nw = gridDim.x * (blockDim.x >> 5);
  const int gtid = blockIdx.x * blockDim.x + threadIdx.x;
  const int nthr = gridDim.x * blockDim.x;
  unsigned rnd = 0;
  for (int s = 0; s < T_; ++s) {
    const int tt = T_ - 1 - s;
    const bf16_t* xr = xe + (size_t)tt * B_ * DX_;
    const bf16_t* dr = dbuf + (size_t)tt * B_ * H_;
    for (int tile = gw; tile < 4 * (G_ / 16); tile += nw) {
      const int tm = (tile & 3) * 16, tn = (tile >> 2) * 16;
      const int m = tm + l15, n = tn + l15;
      const bf16_t* Wr = Wi + (size_t)n * (DX_ + H_);
      v8f ai = wmma_kloop_bb(xr + (size_t)m * DX_, Wr, DX_, v8f_zero(), half);
      ai = wmma_kloop_bb(dr + (size_t)m * H_, Wr + DX_, H_, ai, half);
      v8f ah = wmma_kloop_bb(hbf + (size_t)m * H_, Wh + (size_t)n * H_, H_,
                             v8f_zero(), half);
#pragma unroll
      for (int r = 0; r < 8; ++r) {
        int row = tm + r + 8 * half, col = tn + l15;
        gi[(size_t)row * G_ + col] = ai[r] + bi[col];
        gh[(size_t)row * G_ + col] = ah[r] + bh[col];
      }
    }
    grid_sync(bar, (++rnd) * gridDim.x);
    gru_gate_phase(gi, gh, h, hbf, abuf + (size_t)tt * B_ * H_, gtid, nthr);
    grid_sync(bar, (++rnd) * gridDim.x);
  }
}

// ---------------------------------------------------------------------------
// Kernel 4: latent scan (persistent). Per step: prior/enc MLP3 + 4 Gaussian
// heads + reparameterized z update. 5 grid barriers per step.
// ---------------------------------------------------------------------------
__global__ __launch_bounds__(256) void latent_kernel(
    const bf16_t* __restrict__ dbuf, const bf16_t* __restrict__ abuf,
    const float* __restrict__ eps,
    const bf16_t* pW0, const bf16_t* pW1, const bf16_t* pW2,
    const float* pb0, const float* pb1, const float* pb2,
    const bf16_t* pmW, const float* pmb, const bf16_t* psW, const float* psb,
    const bf16_t* eW0, const bf16_t* eW1, const bf16_t* eW2,
    const float* eb0, const float* eb1, const float* eb2,
    const bf16_t* qmW, const float* qmb, const bf16_t* qsW, const float* qsb,
    bf16_t* hP0, bf16_t* hP1, bf16_t* hE0, bf16_t* hE1, bf16_t* zcur,
    bf16_t* zb, float* pmB, float* psB, float* qmB, float* qsB,
    unsigned* __restrict__ bar) {
  const int lane = threadIdx.x & 31, half = lane >> 4, l15 = lane & 15;
  const int gw = blockIdx.x * (blockDim.x >> 5) + (threadIdx.x >> 5);
  const int nw = gridDim.x * (blockDim.x >> 5);
  const int gtid = blockIdx.x * blockDim.x + threadIdx.x;
  const int nthr = gridDim.x * blockDim.x;
  unsigned rnd = 0;
  for (int t = 0; t < T_; ++t) {
    const bf16_t* drow = dbuf + (size_t)t * B_ * H_;
    const bf16_t* arow = abuf + (size_t)t * B_ * H_;
    // L1: input concat([h_or_a, z]) -> 512, relu. 2 sides x 128 tiles.
    for (int tile = gw; tile < 256; tile += nw) {
      int side = tile >> 7, ti = tile & 127;
      int tm = (ti & 3) * 16, tn = (ti >> 2) * 16;
      int m = tm + l15, n = tn + l15;
      const bf16_t* A0 = (side ? arow : drow) + (size_t)m * H_;
      const bf16_t* W0 = (side ? eW0 : pW0) + (size_t)n * (H_ + Z_);
      const float* b0 = side ? eb0 : pb0;
      bf16_t* O = side ? hE0 : hP0;
      v8f acc = wmma_kloop_bb(A0, W0, H_, v8f_zero(), half);
      acc = wmma_kloop_bb(zcur + (size_t)m * Z_, W0 + H_, Z_, acc, half);
#pragma unroll
      for (int r = 0; r < 8; ++r) {
        int row = tm + r + 8 * half, col = tn + l15;
        O[(size_t)row * H_ + col] = (bf16_t)reluf(acc[r] + b0[col]);
      }
    }
    grid_sync(bar, (++rnd) * gridDim.x);
    // L2: hX0 -> hX1
    for (int tile = gw; tile < 256; tile += nw) {
      int side = tile >> 7, ti = tile & 127;
      int tm = (ti & 3) * 16, tn = (ti >> 2) * 16;
      int m = tm + l15, n = tn + l15;
      const bf16_t* A = (side ? hE0 : hP0) + (size_t)m * H_;
      const bf16_t* W = (side ? eW1 : pW1) + (size_t)n * H_;
      const float* bb = side ? eb1 : pb1;
      bf16_t* O = side ? hE1 : hP1;
      v8f acc = wmma_kloop_bb(A, W, H_, v8f_zero(), half);
#pragma unroll
      for (int r = 0; r < 8; ++r) {
        int row = tm + r + 8 * half, col = tn + l15;
        O[(size_t)row * H_ + col] = (bf16_t)reluf(acc[r] + bb[col]);
      }
    }
    grid_sync(bar, (++rnd) * gridDim.x);
    // L3: hX1 -> hX0
    for (int tile = gw; tile < 256; tile += nw) {
      int side = tile >> 7, ti = tile & 127;
      int tm = (ti & 3) * 16, tn = (ti >> 2) * 16;
      int m = tm + l15, n = tn + l15;
      const bf16_t* A = (side ? hE1 : hP1) + (size_t)m * H_;
      const bf16_t* W = (side ? eW2 : pW2) + (size_t)n * H_;
      const float* bb = side ? eb2 : pb2;
      bf16_t* O = side ? hE0 : hP0;
      v8f acc = wmma_kloop_bb(A, W, H_, v8f_zero(), half);
#pragma unroll
      for (int r = 0; r < 8; ++r) {
        int row = tm + r + 8 * half, col = tn + l15;
        O[(size_t)row * H_ + col] = (bf16_t)reluf(acc[r] + bb[col]);
      }
    }
    grid_sync(bar, (++rnd) * gridDim.x);
    // Heads: pm, ps, qm, qs (64x128 each); sd heads get softplus + 1e-6.
    for (int tile = gw; tile < 128; tile += nw) {
      int head = tile >> 5, ti = tile & 31;
      int tm = (ti & 3) * 16, tn = (ti >> 2) * 16;
      int m = tm + l15, n = tn + l15;
      const bf16_t* A = (head < 2 ? hP0 : hE0) + (size_t)m * H_;
      const bf16_t* W = (head == 0 ? pmW : head == 1 ? psW : head == 2 ? qmW : qsW);
      const float* bb = (head == 0 ? pmb : head == 1 ? psb : head == 2 ? qmb : qsb);
      float* O = (head == 0 ? pmB : head == 1 ? psB : head == 2 ? qmB : qsB);
      v8f acc = wmma_kloop_bb(A, W + (size_t)n * H_, H_, v8f_zero(), half);
      const bool sd = (head & 1);
#pragma unroll
      for (int r = 0; r < 8; ++r) {
        int row = tm + r + 8 * half, col = tn + l15;
        float v = acc[r] + bb[col];
        if (sd) v = softplusf_(v) + 1e-6f;
        O[((size_t)t * B_ + row) * Z_ + col] = v;
      }
    }
    grid_sync(bar, (++rnd) * gridDim.x);
    // z_new = qm + qs * eps[:, t, :]  (fp32 math, bf16 GEMM copies)
    for (int idx = gtid; idx < B_ * Z_; idx += nthr) {
      int b = idx >> 7, k = idx & (Z_ - 1);
      size_t o = ((size_t)t * B_ + b) * Z_ + k;
      float zn = qmB[o] + qsB[o] * eps[((size_t)b * T_ + t) * Z_ + k];
      zcur[idx] = (bf16_t)zn;
      zb[o] = (bf16_t)zn;
    }
    grid_sync(bar, (++rnd) * gridDim.x);
  }
}

// ---------------------------------------------------------------------------
// Kernel 5: decoder MLP3 + likelihood heads + masked log_prob/KL partials.
// One block per 16 rows (same b, consecutive t). bf16 LDS ping-pong.
// ---------------------------------------------------------------------------
__global__ __launch_bounds__(256) void decoder_kernel(
    const float* __restrict__ x, const int* __restrict__ x_sl,
    const bf16_t* __restrict__ zb, const bf16_t* __restrict__ dbuf,
    const float* __restrict__ pmB, const float* __restrict__ psB,
    const float* __restrict__ qmB, const float* __restrict__ qsB,
    const bf16_t* W0, const bf16_t* W1, const bf16_t* W2,
    const float* b0, const float* b1, const float* b2,
    const bf16_t* muW, const float* mub, const bf16_t* sdW, const float* sdb,
    float* __restrict__ partials) {
  __shared__ bf16_t sA[16 * H_];
  __shared__ bf16_t sB[16 * H_];
  __shared__ float sHead[2 * 16 * DX_];
  __shared__ float red[256];
  const int lane = threadIdx.x & 31, half = lane >> 4, l15 = lane & 15;
  const int w = threadIdx.x >> 5;       // wave in block, 0..7
  const int m0 = blockIdx.x * 16;       // row = b*T + t
  const int b = m0 >> 9;
  const int t0 = m0 & (T_ - 1);
  const bf16_t* Az = zb + ((size_t)(t0 + l15) * B_ + b) * Z_;    // [T,B,Z]
  const bf16_t* Ad = dbuf + ((size_t)(t0 + l15) * B_ + b) * H_;  // [T,B,H]
  // L1: concat([z, d]) @ W0^T, relu -> sA
  for (int tn = w; tn < 32; tn += 8) {
    int n = tn * 16 + l15;
    const bf16_t* Wr = W0 + (size_t)n * (Z_ + H_);
    v8f acc = wmma_kloop_bb(Az, Wr, Z_, v8f_zero(), half);
    acc = wmma_kloop_bb(Ad, Wr + Z_, H_, acc, half);
#pragma unroll
    for (int r = 0; r < 8; ++r)
      sA[(size_t)(r + 8 * half) * H_ + n] = (bf16_t)reluf(acc[r] + b0[n]);
  }
  __syncthreads();
  // L2: sA -> sB
  for (int tn = w; tn < 32; tn += 8) {
    int n = tn * 16 + l15;
    v8f acc = wmma_kloop_bb(sA + (size_t)l15 * H_, W1 + (size_t)n * H_, H_,
                            v8f_zero(), half);
#pragma unroll
    for (int r = 0; r < 8; ++r)
      sB[(size_t)(r + 8 * half) * H_ + n] = (bf16_t)reluf(acc[r] + b1[n]);
  }
  __syncthreads();
  // L3: sB -> sA
  for (int tn = w; tn < 32; tn += 8) {
    int n = tn * 16 + l15;
    v8f acc = wmma_kloop_bb(sB + (size_t)l15 * H_, W2 + (size_t)n * H_, H_,
                            v8f_zero(), half);
#pragma unroll
    for (int r = 0; r < 8; ++r)
      sA[(size_t)(r + 8 * half) * H_ + n] = (bf16_t)reluf(acc[r] + b2[n]);
  }
  __syncthreads();
  // Heads: xm -> sHead[0:2048], xsd -> sHead[2048:4096]  (fp32)
  {
    int n = w * 16 + l15;
    v8f am = wmma_kloop_bb(sA + (size_t)l15 * H_, muW + (size_t)n * H_, H_,
                           v8f_zero(), half);
    v8f as = wmma_kloop_bb(sA + (size_t)l15 * H_, sdW + (size_t)n * H_, H_,
                           v8f_zero(), half);
#pragma unroll
    for (int r = 0; r < 8; ++r) {
      int row = r + 8 * half;
      sHead[(size_t)row * DX_ + n] = am[r] + mub[n];
      sHead[2048 + (size_t)row * DX_ + n] = softplusf_(as[r] + sdb[n]) + 1e-6f;
    }
  }
  __syncthreads();
  // Masked log-likelihood + KL partial sums for this block's 16 rows.
  const int sl = x_sl[b];
  float lp = 0.0f, kl = 0.0f;
  for (int idx = threadIdx.x; idx < 16 * DX_; idx += 256) {
    int rr = idx >> 7, c = idx & (DX_ - 1);
    int t = t0 + rr;
    if (t < sl) {
      float xm = sHead[idx], xsd = sHead[2048 + idx];
      float yv = x[((size_t)b * T_ + t) * DX_ + c];
      float dv = (yv - xm) / xsd;
      lp += -0.5f * dv * dv - logf(xsd) - 0.9189385332046727f;
      size_t o = ((size_t)t * B_ + b) * Z_ + c;
      float pmv = pmB[o], psv = psB[o], qmv = qmB[o], qsv = qsB[o];
      float dm = qmv - pmv;
      kl += logf(psv) - logf(qsv) +
            (qsv * qsv + dm * dm) / (2.0f * psv * psv) - 0.5f;
    }
  }
  red[threadIdx.x] = lp;
  __syncthreads();
  for (int s = 128; s > 0; s >>= 1) {
    if (threadIdx.x < s) red[threadIdx.x] += red[threadIdx.x + s];
    __syncthreads();
  }
  if (threadIdx.x == 0) partials[blockIdx.x] = red[0];
  __syncthreads();
  red[threadIdx.x] = kl;
  __syncthreads();
  for (int s = 128; s > 0; s >>= 1) {
    if (threadIdx.x < s) red[threadIdx.x] += red[threadIdx.x + s];
    __syncthreads();
  }
  if (threadIdx.x == 0) partials[NROWTILE + blockIdx.x] = red[0];
}

// Kernel 6: deterministic final reduction -> scalar loss.
__global__ __launch_bounds__(256) void final_kernel(
    const float* __restrict__ partials, const int* __restrict__ x_sl,
    float* __restrict__ out) {
  __shared__ float rl[256], rk[256];
  __shared__ int ri[256];
  float lp = 0.0f, kl = 0.0f;
  int sl = 0;
  for (int i = threadIdx.x; i < NROWTILE; i += 256) {
    lp += partials[i];
    kl += partials[NROWTILE + i];
  }
  if (threadIdx.x < B_) sl = x_sl[threadIdx.x];
  rl[threadIdx.x] = lp; rk[threadIdx.x] = kl; ri[threadIdx.x] = sl;
  __syncthreads();
  for (int s = 128; s > 0; s >>= 1) {
    if (threadIdx.x < s) {
      rl[threadIdx.x] += rl[threadIdx.x + s];
      rk[threadIdx.x] += rk[threadIdx.x + s];
      ri[threadIdx.x] += ri[threadIdx.x + s];
    }
    __syncthreads();
  }
  if (threadIdx.x == 0) out[0] = -(rl[0] - rk[0]) / (float)ri[0];
}

// ---------------------------------------------------------------------------
extern "C" void kernel_launch(void* const* d_in, const int* in_sizes, int n_in,
                              void* d_out, int out_size, void* d_ws,
                              size_t ws_size, hipStream_t stream) {
  (void)in_sizes; (void)n_in; (void)out_size; (void)ws_size;
  const float* x    = (const float*)d_in[0];
  const int*   x_sl = (const int*)d_in[1];
  const float* eps  = (const float*)d_in[2];
  // fp32 parameter pointers (biases used directly; weights get converted).
  const float* P[40];
  for (int i = 0; i < 40; ++i) P[i] = (const float*)d_in[3 + i];
  // P index map (see flattening comment at top, minus the first 3 inputs):
  //  0:emb_W 1:emb_b 2:gruf_Wi 3:gruf_Wh 4:gruf_bi 5:gruf_bh
  //  6:grub_Wi 7:grub_Wh 8:grub_bi 9:grub_bh
  //  10..12:pW0..2 13..15:pb0..2 16:pmW 17:pmb 18:psW 19:psb
  //  20..22:eW0..2 23..25:eb0..2 26:qmW 27:qmb 28:qsW 29:qsb
  //  30..32:dW0..2 33..35:db0..2 36:likmuW 37:likmub 38:liksdW 39:liksdb

  // ---- workspace carve-up: fp32 region first, then 16B-aligned bf16 region
  float* ws = (float*)d_ws;
  size_t off = 0;
  auto allocf = [&](size_t n) { float* p = ws + off; off += n; return p; };
  float* giS  = allocf((size_t)B_ * G_);
  float* ghS  = allocf((size_t)B_ * G_);
  float* hS   = allocf((size_t)B_ * H_);
  float* pmB  = allocf((size_t)T_ * B_ * Z_);
  float* psB  = allocf((size_t)T_ * B_ * Z_);
  float* qmB  = allocf((size_t)T_ * B_ * Z_);
  float* qsB  = allocf((size_t)T_ * B_ * Z_);
  float* partials = allocf(2 * NROWTILE);
  unsigned* bars = (unsigned*)(ws + off); off += 16;
  off = (off + 3) & ~(size_t)3;  // 16B align for bf16 region

  bf16_t* wb = (bf16_t*)(ws + off);
  size_t boff = 0;
  auto allocb = [&](size_t n) { bf16_t* p = wb + boff; boff += n; return p; };
  bf16_t* xe   = allocb((size_t)T_ * B_ * DX_);
  bf16_t* dbuf = allocb((size_t)T_ * B_ * H_);
  bf16_t* abuf = allocb((size_t)T_ * B_ * H_);
  bf16_t* zb   = allocb((size_t)T_ * B_ * Z_);
  bf16_t* hbf  = allocb((size_t)B_ * H_);
  bf16_t* hP0  = allocb((size_t)B_ * H_);
  bf16_t* hP1  = allocb((size_t)B_ * H_);
  bf16_t* hE0  = allocb((size_t)B_ * H_);
  bf16_t* hE1  = allocb((size_t)B_ * H_);
  bf16_t* zcur = allocb((size_t)B_ * Z_);

  // bf16 weight arena: convert the 20 GEMM weight matrices once.
  const int widx[20] = {0, 2, 3, 6, 7, 10, 11, 12, 16, 18,
                        20, 21, 22, 26, 28, 30, 31, 32, 36, 38};
  const size_t wsz[20] = {
      (size_t)DX_ * DX_,            // emb_W
      (size_t)G_ * DX_,             // gruf_Wi
      (size_t)G_ * H_,              // gruf_Wh
      (size_t)G_ * (DX_ + H_),      // grub_Wi
      (size_t)G_ * H_,              // grub_Wh
      (size_t)H_ * (H_ + Z_), (size_t)H_ * H_, (size_t)H_ * H_,  // pW0..2
      (size_t)Z_ * H_, (size_t)Z_ * H_,                           // pmW, psW
      (size_t)H_ * (H_ + Z_), (size_t)H_ * H_, (size_t)H_ * H_,  // eW0..2
      (size_t)Z_ * H_, (size_t)Z_ * H_,                           // qmW, qsW
      (size_t)H_ * (H_ + Z_), (size_t)H_ * H_, (size_t)H_ * H_,  // dW0..2
      (size_t)DX_ * H_, (size_t)DX_ * H_};                        // lik mu/sd
  bf16_t* WB[20];
  for (int i = 0; i < 20; ++i) WB[i] = allocb(wsz[i]);

  hipMemsetAsync(bars, 0, 16 * sizeof(unsigned), stream);
  for (int i = 0; i < 20; ++i) {
    int n = (int)wsz[i];
    f2bf_kernel<<<(n + 255) / 256, 256, 0, stream>>>(P[widx[i]], WB[i], n);
  }
  bf16_t* emb_W   = WB[0];
  bf16_t* gruf_Wi = WB[1];  bf16_t* gruf_Wh = WB[2];
  bf16_t* grub_Wi = WB[3];  bf16_t* grub_Wh = WB[4];
  bf16_t* pW0 = WB[5];  bf16_t* pW1 = WB[6];  bf16_t* pW2 = WB[7];
  bf16_t* pmW = WB[8];  bf16_t* psW = WB[9];
  bf16_t* eW0 = WB[10]; bf16_t* eW1 = WB[11]; bf16_t* eW2 = WB[12];
  bf16_t* qmW = WB[13]; bf16_t* qsW = WB[14];
  bf16_t* dW0 = WB[15]; bf16_t* dW1 = WB[16]; bf16_t* dW2 = WB[17];
  bf16_t* likmuW = WB[18]; bf16_t* liksdW = WB[19];

  // 1) embedding
  emb_kernel<<<NROWTILE, 256, 0, stream>>>(x, emb_W, P[1], xe);
  // 2) forward GRU (h0 = 0)
  hipMemsetAsync(hS, 0, (size_t)B_ * H_ * sizeof(float), stream);
  hipMemsetAsync(hbf, 0, (size_t)B_ * H_ * sizeof(bf16_t), stream);
  gru_fwd_kernel<<<NBLK_SCAN, 256, 0, stream>>>(xe, gruf_Wi, gruf_Wh, P[4],
                                                P[5], dbuf, hS, hbf, giS, ghS,
                                                &bars[0]);
  // 3) backward GRU (h0 = 0)
  hipMemsetAsync(hS, 0, (size_t)B_ * H_ * sizeof(float), stream);
  hipMemsetAsync(hbf, 0, (size_t)B_ * H_ * sizeof(bf16_t), stream);
  gru_bwd_kernel<<<NBLK_SCAN, 256, 0, stream>>>(xe, dbuf, grub_Wi, grub_Wh,
                                                P[8], P[9], abuf, hS, hbf, giS,
                                                ghS, &bars[1]);
  // 4) latent scan (z0 = 0)
  hipMemsetAsync(zcur, 0, (size_t)B_ * Z_ * sizeof(bf16_t), stream);
  latent_kernel<<<NBLK_SCAN, 256, 0, stream>>>(
      dbuf, abuf, eps, pW0, pW1, pW2, P[13], P[14], P[15], pmW, P[17], psW,
      P[19], eW0, eW1, eW2, P[23], P[24], P[25], qmW, P[27], qsW, P[29], hP0,
      hP1, hE0, hE1, zcur, zb, pmB, psB, qmB, qsB, &bars[2]);
  // 5) decoder + likelihood + masked partial sums
  decoder_kernel<<<NROWTILE, 256, 0, stream>>>(
      x, x_sl, zb, dbuf, pmB, psB, qmB, qsB, dW0, dW1, dW2, P[33], P[34],
      P[35], likmuW, P[36], liksdW, P[39], partials);
  // 6) scalar loss
  final_kernel<<<1, 256, 0, stream>>>(partials, x_sl, (float*)d_out);
}
